// MultiHeadAttention_90426241450797
// MI455X (gfx1250) — compile-verified, matrix operand order checked by reading, not run on hardware
//
#include <hip/hip_runtime.h>

#define DEV __device__ __forceinline__

typedef _Float16 half_t;
typedef __attribute__((ext_vector_type(16))) _Float16 v16h;
typedef __attribute__((ext_vector_type(8)))  _Float16 v8h;
typedef __attribute__((ext_vector_type(8)))  float    v8f;

constexpr int Bb   = 2;
constexpr int Ss   = 2048;
constexpr int Dd   = 1024;
constexpr int Hh   = 16;
constexpr int HD   = 64;
constexpr int MTOK = Bb * Ss;   // 4096 tokens

// ---------------------------------------------------------------------------
// Fragment loader for 16-bit A/B WMMA operands (row-major source).
// ISA layout: lanes 0-15 hold K chunks {0..7,16..23}; lanes 16-31 hold
// {8..15,24..31}; caller passes the two absolute K offsets (incl. hi*8).
// Each half is one 16-byte global_load_b128 / ds_load_b128.
// ---------------------------------------------------------------------------
DEV v16h ldfrag(const half_t* __restrict__ rowp, int ka, int kb) {
  v8h x = *(const v8h*)(rowp + ka);
  v8h y = *(const v8h*)(rowp + kb);
  v16h r;
#pragma unroll
  for (int i = 0; i < 8; ++i) { r[i] = x[i]; r[8 + i] = y[i]; }
  return r;
}

DEV v8f wmma_f16(v16h a, v16h b, v8f c) {
  return __builtin_amdgcn_wmma_f32_16x16x32_f16(
      false, a, false, b, (short)0, c, false, false);
}

// Row-wise reductions across the 16 lanes sharing one row set of the C-frag.
DEV float rmax16(float v) {
#pragma unroll
  for (int m = 1; m <= 8; m <<= 1) v = fmaxf(v, __shfl_xor(v, m, 32));
  return v;
}
DEV float rsum16(float v) {
#pragma unroll
  for (int m = 1; m <= 8; m <<= 1) v += __shfl_xor(v, m, 32);
  return v;
}

// ---------------------------------------------------------------------------
// Precision conversion kernels
// ---------------------------------------------------------------------------
__global__ __launch_bounds__(256) void cvt_f32_to_f16(
    const float* __restrict__ src, half_t* __restrict__ dst, int n) {
  int i = blockIdx.x * blockDim.x + threadIdx.x;
  if (i < n) dst[i] = (half_t)src[i];
}

// W [K][N] fp32 -> Wt [N][K] f16 so B-fragments are contiguous loads.
__global__ __launch_bounds__(256) void transpose_to_f16(
    const float* __restrict__ W, half_t* __restrict__ Wt) {
  int idx = blockIdx.x * blockDim.x + threadIdx.x;   // dst linear: n*Dd + k
  int k = idx & (Dd - 1);
  int n = idx >> 10;
  Wt[idx] = (half_t)W[(size_t)k * Dd + n];
}

// ---------------------------------------------------------------------------
// Fused QKV projection: one wave = one 16x16 tile of Q, K and V.
// Double-buffered K-loop: iteration k+1 fragments are in flight while the
// WMMAs of iteration k execute (wait becomes s_wait_loadcnt N, not 0).
// ---------------------------------------------------------------------------
__global__ __launch_bounds__(32) void qkv_kernel(
    const half_t* __restrict__ Ah,
    const half_t* __restrict__ WqT, const half_t* __restrict__ WkT,
    const half_t* __restrict__ WvT,
    const float* __restrict__ bq, const float* __restrict__ bk,
    const float* __restrict__ bv,
    half_t* __restrict__ Qh, half_t* __restrict__ Kh, half_t* __restrict__ Vt) {
  const int lane = threadIdx.x;
  const int cn = lane & 15;
  const int hi = lane >> 4;
  const int m0 = blockIdx.x * 16;
  const int n0 = blockIdx.y * 16;
  const int ka = hi * 8, kb = 16 + hi * 8;

  const half_t* arow = Ah  + (size_t)(m0 + cn) * Dd;
  const half_t* wq   = WqT + (size_t)(n0 + cn) * Dd;
  const half_t* wk   = WkT + (size_t)(n0 + cn) * Dd;
  const half_t* wv   = WvT + (size_t)(n0 + cn) * Dd;

  v8f cq = {}, ck = {}, cv = {};

  // prologue: fragments for k0 = 0
  v16h a  = ldfrag(arow, ka, kb);
  v16h b0 = ldfrag(wq,   ka, kb);
  v16h b1 = ldfrag(wk,   ka, kb);
  v16h b2 = ldfrag(wv,   ka, kb);

  for (int k0 = 0; k0 < Dd; k0 += 32) {
    const int kn = (k0 + 32 < Dd) ? k0 + 32 : 0;   // last iter: harmless reload
    v16h an  = ldfrag(arow, kn + ka, kn + kb);
    v16h b0n = ldfrag(wq,   kn + ka, kn + kb);
    v16h b1n = ldfrag(wk,   kn + ka, kn + kb);
    v16h b2n = ldfrag(wv,   kn + ka, kn + kb);

    cq = wmma_f16(a, b0, cq);
    ck = wmma_f16(a, b1, ck);
    cv = wmma_f16(a, b2, cv);

    a = an; b0 = b0n; b1 = b1n; b2 = b2n;
  }

  const float bqv = bq[n0 + cn];
  const float bkv = bk[n0 + cn];
  const float bvv = bv[n0 + cn];
  const int h     = n0 >> 6;
  const int dcol  = (n0 & 63) + cn;
  const int b     = m0 >> 11;
  const int sbase = m0 & (Ss - 1);

#pragma unroll
  for (int i = 0; i < 8; ++i) {
    int s = sbase + i + 8 * hi;
    size_t base = ((size_t)(b * Hh + h) * Ss + s) * HD + dcol;
    Qh[base] = (half_t)((cq[i] + bqv) * 0.125f);   // fold 1/sqrt(64)
    Kh[base] = (half_t)(ck[i] + bkv);
  }
  v8h vv;
#pragma unroll
  for (int i = 0; i < 8; ++i) vv[i] = (half_t)(cv[i] + bvv);
  *(v8h*)(Vt + ((size_t)(b * Hh + h) * HD + dcol) * Ss + sbase + 8 * hi) = vv;
}

// ---------------------------------------------------------------------------
// Causal flash attention. One wave per (b, h, 16-query tile).
// V fragments are loaded at chunk start (consumed only after softmax, so
// their latency is hidden); K fragments for the next chunk are prefetched
// into registers while the current chunk's scores/softmax execute.
// ---------------------------------------------------------------------------
__global__ __launch_bounds__(32) void attn_kernel(
    const half_t* __restrict__ Qh, const half_t* __restrict__ Kh,
    const half_t* __restrict__ Vt, half_t* __restrict__ AOut) {
  __shared__ half_t Plds[16 * 32];

  const int lane = threadIdx.x;
  const int cn = lane & 15;
  const int hi = lane >> 4;
  const int q0 = blockIdx.x * 16;
  const int bh = blockIdx.y;
  const int b  = bh >> 4;
  const int h  = bh & 15;

  const half_t* Qp = Qh + (size_t)bh * Ss * HD;
  const half_t* Kp = Kh + (size_t)bh * Ss * HD;
  const half_t* Vp = Vt + (size_t)bh * HD * Ss;

  const half_t* qrow = Qp + (size_t)(q0 + cn) * HD;
  const v16h aq0 = ldfrag(qrow, hi * 8,      16 + hi * 8);
  const v16h aq1 = ldfrag(qrow, 32 + hi * 8, 48 + hi * 8);

  float mrow[8], lrow[8];
#pragma unroll
  for (int i = 0; i < 8; ++i) { mrow[i] = -1e30f; lrow[i] = 0.f; }
  v8f o0 = {}, o1 = {}, o2 = {}, o3 = {};

  auto loadK = [&](int kvbase, v16h& c0lo, v16h& c0hi, v16h& c1lo, v16h& c1hi) {
    int kr0 = kvbase + cn;      if (kr0 > Ss - 1) kr0 = Ss - 1;   // clamped rows
    int kr1 = kvbase + 16 + cn; if (kr1 > Ss - 1) kr1 = Ss - 1;   // are masked
    const half_t* r0 = Kp + (size_t)kr0 * HD;
    const half_t* r1 = Kp + (size_t)kr1 * HD;
    c0lo = ldfrag(r0, hi * 8, 16 + hi * 8);
    c0hi = ldfrag(r0, 32 + hi * 8, 48 + hi * 8);
    c1lo = ldfrag(r1, hi * 8, 16 + hi * 8);
    c1hi = ldfrag(r1, 32 + hi * 8, 48 + hi * 8);
  };

  const int kv_end = q0 + 15;
  v16h k0lo, k0hi, k1lo, k1hi;
  loadK(0, k0lo, k0hi, k1lo, k1hi);

  for (int kv0 = 0; kv0 <= kv_end; kv0 += 32) {
    // ---- V fragments for this chunk: issued ~30 instrs before use ----
    int ca = kv0 + hi * 8;      if (ca > Ss - 8) ca = Ss - 8;   // clamp; P==0
    int cb = kv0 + 16 + hi * 8; if (cb > Ss - 8) cb = Ss - 8;
    v16h bv0 = ldfrag(Vp + (size_t)( 0 + cn) * Ss, ca, cb);
    v16h bv1 = ldfrag(Vp + (size_t)(16 + cn) * Ss, ca, cb);
    v16h bv2 = ldfrag(Vp + (size_t)(32 + cn) * Ss, ca, cb);
    v16h bv3 = ldfrag(Vp + (size_t)(48 + cn) * Ss, ca, cb);

    // ---- prefetch next chunk's K fragments into registers ----
    const int kvn = (kv0 + 32 <= kv_end) ? kv0 + 32 : kv0;
    v16h nk0lo, nk0hi, nk1lo, nk1hi;
    loadK(kvn, nk0lo, nk0hi, nk1lo, nk1hi);
    if (kv0 + 64 <= kv_end)
      __builtin_prefetch(Kp + (size_t)(kv0 + 64 + cn) * HD, 0, 0);

    // ---- scores: two 16x16 subtiles over the 32-wide KV chunk ----
    v8f s0 = {}, s1 = {};
    s0 = wmma_f16(aq0, k0lo, s0);
    s0 = wmma_f16(aq1, k0hi, s0);
    s1 = wmma_f16(aq0, k1lo, s1);
    s1 = wmma_f16(aq1, k1hi, s1);

    // ---- causal mask + online softmax ----
    float p0[8], p1[8], alpha[8];
#pragma unroll
    for (int i = 0; i < 8; ++i) {
      int qi = q0 + i + 8 * hi;
      if (kv0 + cn      > qi) s0[i] = -1e30f;
      if (kv0 + 16 + cn > qi) s1[i] = -1e30f;
      float v  = rmax16(fmaxf(s0[i], s1[i]));
      float mn = fmaxf(mrow[i], v);
      alpha[i] = __expf(mrow[i] - mn);
      mrow[i]  = mn;
      p0[i] = __expf(s0[i] - mn);
      p1[i] = __expf(s1[i] - mn);
      lrow[i] = lrow[i] * alpha[i] + rsum16(p0[i] + p1[i]);
    }
#pragma unroll
    for (int i = 0; i < 8; ++i) {
      o0[i] *= alpha[i]; o1[i] *= alpha[i];
      o2[i] *= alpha[i]; o3[i] *= alpha[i];
    }

    // ---- P: C-layout f32 -> LDS f16 -> A-layout fragment ----
#pragma unroll
    for (int i = 0; i < 8; ++i) {
      int row = i + 8 * hi;
      Plds[row * 32 + cn]      = (half_t)p0[i];
      Plds[row * 32 + 16 + cn] = (half_t)p1[i];
    }
    asm volatile("s_wait_dscnt 0" ::: "memory");   // CDNA5 split DS counter
    v16h ap;
#pragma unroll
    for (int i = 0; i < 8; ++i) {
      ap[i]     = Plds[cn * 32 + hi * 8 + i];
      ap[8 + i] = Plds[cn * 32 + 16 + hi * 8 + i];
    }

    // ---- P x V (V data long since arrived) ----
    o0 = wmma_f16(ap, bv0, o0);
    o1 = wmma_f16(ap, bv1, o1);
    o2 = wmma_f16(ap, bv2, o2);
    o3 = wmma_f16(ap, bv3, o3);

    k0lo = nk0lo; k0hi = nk0hi; k1lo = nk1lo; k1hi = nk1hi;
  }

  // ---- normalize, store attention output f16 at [token, h*64 + d] ----
#pragma unroll
  for (int i = 0; i < 8; ++i) {
    float inv = 1.0f / lrow[i];
    int tok = b * Ss + q0 + i + 8 * hi;
    half_t* dst = AOut + (size_t)tok * Dd + h * HD;
    dst[ 0 + cn] = (half_t)(o0[i] * inv);
    dst[16 + cn] = (half_t)(o1[i] * inv);
    dst[32 + cn] = (half_t)(o2[i] * inv);
    dst[48 + cn] = (half_t)(o3[i] * inv);
  }
}

// ---------------------------------------------------------------------------
// Output projection: 16x64 stripe per wave (A-fragment reused over 4 N-tiles),
// double-buffered K-loop. AttnOut(f16) x Wo^T(f16) + bo -> fp32.
// ---------------------------------------------------------------------------
__global__ __launch_bounds__(32) void oproj_kernel(
    const half_t* __restrict__ Ah, const half_t* __restrict__ WoT,
    const float* __restrict__ bo, float* __restrict__ Out) {
  const int lane = threadIdx.x;
  const int cn = lane & 15;
  const int hi = lane >> 4;
  const int m0 = blockIdx.x * 16;
  const int n0 = blockIdx.y * 64;
  const int ka = hi * 8, kb = 16 + hi * 8;

  const half_t* arow = Ah + (size_t)(m0 + cn) * Dd;
  const half_t* w0 = WoT + (size_t)(n0 +  0 + cn) * Dd;
  const half_t* w1 = WoT + (size_t)(n0 + 16 + cn) * Dd;
  const half_t* w2 = WoT + (size_t)(n0 + 32 + cn) * Dd;
  const half_t* w3 = WoT + (size_t)(n0 + 48 + cn) * Dd;

  v8f c0 = {}, c1 = {}, c2 = {}, c3 = {};

  v16h a  = ldfrag(arow, ka, kb);
  v16h b0 = ldfrag(w0,   ka, kb);
  v16h b1 = ldfrag(w1,   ka, kb);
  v16h b2 = ldfrag(w2,   ka, kb);
  v16h b3 = ldfrag(w3,   ka, kb);

  for (int k0 = 0; k0 < Dd; k0 += 32) {
    const int kn = (k0 + 32 < Dd) ? k0 + 32 : 0;
    v16h an  = ldfrag(arow, kn + ka, kn + kb);
    v16h b0n = ldfrag(w0,   kn + ka, kn + kb);
    v16h b1n = ldfrag(w1,   kn + ka, kn + kb);
    v16h b2n = ldfrag(w2,   kn + ka, kn + kb);
    v16h b3n = ldfrag(w3,   kn + ka, kn + kb);

    c0 = wmma_f16(a, b0, c0);
    c1 = wmma_f16(a, b1, c1);
    c2 = wmma_f16(a, b2, c2);
    c3 = wmma_f16(a, b3, c3);

    a = an; b0 = b0n; b1 = b1n; b2 = b2n; b3 = b3n;
  }

  const float bi0 = bo[n0 +  0 + cn];
  const float bi1 = bo[n0 + 16 + cn];
  const float bi2 = bo[n0 + 32 + cn];
  const float bi3 = bo[n0 + 48 + cn];
#pragma unroll
  for (int i = 0; i < 8; ++i) {
    float* orow = Out + (size_t)(m0 + i + 8 * hi) * Dd + n0;
    orow[ 0 + cn] = c0[i] + bi0;
    orow[16 + cn] = c1[i] + bi1;
    orow[32 + cn] = c2[i] + bi2;
    orow[48 + cn] = c3[i] + bi3;
  }
}

// ---------------------------------------------------------------------------
// Host launcher. Workspace layout (40 MB):
//   [0,8M)    q f16 (reused as AttnOut f16 after QKV GEMM)
//   [8M,16M)  WqT/WkT/WvT/WoT f16 (2 MB each)
//   [16M,24M) Q f16 [B,H,S,64] (pre-scaled by 1/8)
//   [24M,32M) K f16 [B,H,S,64]
//   [32M,40M) V^T f16 [B,H,64,S]
// ---------------------------------------------------------------------------
extern "C" void kernel_launch(void* const* d_in, const int* in_sizes, int n_in,
                              void* d_out, int out_size, void* d_ws, size_t ws_size,
                              hipStream_t stream) {
  const float* q  = (const float*)d_in[0];
  const float* Wq = (const float*)d_in[1];
  const float* bq = (const float*)d_in[2];
  const float* Wk = (const float*)d_in[3];
  const float* bk = (const float*)d_in[4];
  const float* Wv = (const float*)d_in[5];
  const float* bv = (const float*)d_in[6];
  const float* Wo = (const float*)d_in[7];
  const float* bo = (const float*)d_in[8];
  float* out = (float*)d_out;

  char* ws = (char*)d_ws;
  const size_t SZ_TOK = (size_t)MTOK * Dd * sizeof(half_t);  // 8 MB
  const size_t SZ_W   = (size_t)Dd * Dd * sizeof(half_t);    // 2 MB
  half_t* qh   = (half_t*)(ws);
  half_t* WqT  = (half_t*)(ws + SZ_TOK);
  half_t* WkT  = (half_t*)(ws + SZ_TOK + 1 * SZ_W);
  half_t* WvT  = (half_t*)(ws + SZ_TOK + 2 * SZ_W);
  half_t* WoT  = (half_t*)(ws + SZ_TOK + 3 * SZ_W);
  half_t* Qh   = (half_t*)(ws + SZ_TOK + 4 * SZ_W);
  half_t* Kh   = (half_t*)(ws + 2 * SZ_TOK + 4 * SZ_W);
  half_t* Vt   = (half_t*)(ws + 3 * SZ_TOK + 4 * SZ_W);
  half_t* AOut = qh;

  {
    int n = MTOK * Dd;
    cvt_f32_to_f16<<<n / 256, 256, 0, stream>>>(q, qh, n);
    int wblk = (Dd * Dd) / 256;
    transpose_to_f16<<<wblk, 256, 0, stream>>>(Wq, WqT);
    transpose_to_f16<<<wblk, 256, 0, stream>>>(Wk, WkT);
    transpose_to_f16<<<wblk, 256, 0, stream>>>(Wv, WvT);
    transpose_to_f16<<<wblk, 256, 0, stream>>>(Wo, WoT);
  }

  qkv_kernel<<<dim3(MTOK / 16, Dd / 16), 32, 0, stream>>>(
      qh, WqT, WkT, WvT, bq, bk, bv, Qh, Kh, Vt);

  attn_kernel<<<dim3(Ss / 16, Bb * Hh), 32, 0, stream>>>(Qh, Kh, Vt, AOut);

  oproj_kernel<<<dim3(MTOK / 16, Dd / 64), 32, 0, stream>>>(AOut, WoT, bo, out);
}
